// SPDecoder_86002425135144
// MI455X (gfx1250) — matrix operand order repaired and checked by reading, not compiled, not run on hardware
//
#include <hip/hip_runtime.h>

// CDNA5 / gfx1250, wave32. Two-phase HGCN decoder:
//   Phase 1 (node-major, WMMA): p_tiled = logmap0(x) @ [W_top | W_bot | 0]
//   Phase 2 (edge-major gather): out[e] = p[src][0:4] + p[dst][4:8] + b
//
// Roofline: restructured so each edge gathers 32 B (from a 6.4 MB L2-resident
// table) instead of 1 KB of h-rows; total traffic ~120 MB @ 23.3 TB/s.

typedef __attribute__((ext_vector_type(16))) _Float16  v16h;
typedef __attribute__((ext_vector_type(8)))  float     v8f;
typedef __attribute__((ext_vector_type(4)))  float     v4f;
typedef __attribute__((ext_vector_type(2)))  long long v2i64;

#define DIM   128
#define NCLS  4

// ---------------------------------------------------------------------------
// Phase 1: one wave32 computes a 16-node x 16-col tile, K = 128 via 4 chained
// v_wmma_f32_16x16x32_f16.  Entirely branch-free inside the wave.
//
// p layout (tiled, matches D VGPR layout): element (node nd, col n) lives at
//   p[(nd>>4)*256 + (nd&7)*32 + ((nd>>3)&1)*16 + n]
// so lane writes c[r] -> p[tile*256 + r*32 + lane]  (coalesced, unconditional).
// ---------------------------------------------------------------------------
__global__ __launch_bounds__(256)
void node_logmap_gemm(const float* __restrict__ x,
                      const float* __restrict__ W,   // [2*DIM, NCLS] row-major
                      float*       __restrict__ p,   // [nTiles*256] tiled
                      int nNodes, int nTiles)
{
    const int lane = threadIdx.x & 31;
    const int wave = threadIdx.x >> 5;
    const int tile = blockIdx.x * 8 + wave;
    if (tile >= nTiles) return;          // uniform per-wave exit: EXEC all-1s below

    const int mn   = lane & 15;          // A row index == B col index
    const int half = lane >> 4;

    // ---- B tiles, branch-free.  B col n: 0..3 = W_top, 4..7 = W_bot, 8..15 = 0.
    // Unified address: W[k*4 + nn + (nn>=4 ? 508 : 0)], masked for n>=8.
    const int   nn    = mn & 7;
    const float wmask = (mn < 8) ? 1.0f : 0.0f;
    const int   cofs  = nn + ((nn >= 4) ? (DIM * NCLS - NCLS) : 0);   // +508
    v16h Bt[4];
    #pragma unroll
    for (int t = 0; t < 4; ++t) {
        #pragma unroll
        for (int j = 0; j < 16; ++j) {
            const int k = t * 32 + (j >> 3) * 16 + half * 8 + (j & 7);       // B: K = t*32 + (j>>3)*16 + half*8 + (j&7)
            Bt[t][j] = (_Float16)(W[k * NCLS + cofs] * wmask);
        }
    }

    // ---- A loads: lane (half, m) owns K = t*32 + {0,16} + half*8 + 0..7.
    // Tail rows: clamp pointer (their D rows land in p slots phase 2 never reads).
    const int node  = tile * 16 + mn;
    const int nodeC = (node < nNodes) ? node : (nNodes - 1);
    const float* row = x + (size_t)nodeC * DIM;

    float vals[64];
    float ss = 0.0f;
    #pragma unroll
    for (int t = 0; t < 4; ++t) {
        #pragma unroll
        for (int g = 0; g < 2; ++g) {
            const int kbase = t * 32 + g * 16 + half * 8;
            const v4f v0 = *(const v4f*)(row + kbase);
            const v4f v1 = *(const v4f*)(row + kbase + 4);
            float* dst = &vals[t * 16 + g * 8];
            #pragma unroll
            for (int u = 0; u < 4; ++u) {
                dst[u]     = v0[u];  ss = fmaf(v0[u], v0[u], ss);
                dst[4 + u] = v1[u];  ss = fmaf(v1[u], v1[u], ss);
            }
        }
    }
    // lanes m and m+16 each hold half of row m's 128 elements
    ss += __shfl_xor(ss, 16, 32);

    float norm  = fmaxf(sqrtf(ss), 1e-15f);      // MIN_NORM clamp
    float a     = fminf(norm, 1.0f - 1e-6f);     // artanh arg clamp (sqrt_c = 1)
    float scale = atanhf(a) / norm;              // logmap0 scale

    v16h At[4];
    #pragma unroll
    for (int t = 0; t < 4; ++t)
        #pragma unroll
        for (int j = 0; j < 16; ++j)
            At[t][j] = (_Float16)(vals[t * 16 + j] * scale);

    // ---- D = sum_t A_t x B_t, f32 accumulate
    v8f c = {};
    #pragma unroll
    for (int t = 0; t < 4; ++t)
        c = __builtin_amdgcn_wmma_f32_16x16x32_f16(
                /*neg_a=*/false, At[t], /*neg_b=*/false, Bt[t],
                /*c_mod=*/(short)0, c, /*reuse_a=*/false, /*reuse_b=*/false);

    // ---- Unconditional coalesced stores in the tiled layout
    float* pt = p + (size_t)tile * 256;
    #pragma unroll
    for (int r = 0; r < 8; ++r)
        pt[r * 32 + lane] = c[r];
}

// ---------------------------------------------------------------------------
// Phase 2: 32 B gather per edge from L2-resident p; idx/out are one-shot
// streams -> nontemporal so they don't evict p from L2.  Node ids < 2^31, so
// all tiled-address math is done in 32-bit (saddr + voffset addressing).
// ---------------------------------------------------------------------------
__global__ __launch_bounds__(256)
void edge_decode(const float*     __restrict__ p,     // tiled layout (see above)
                 const long long* __restrict__ idx,   // [nEdges, 2], int64
                 const float*     __restrict__ b,     // [NCLS]
                 float*           __restrict__ out,   // [nEdges, NCLS]
                 int nEdges)
{
    const int e = blockIdx.x * blockDim.x + threadIdx.x;
    if (e >= nEdges) return;

    const v2i64 sd = __builtin_nontemporal_load((const v2i64*)(idx + 2 * (size_t)e));
    const int s = (int)sd[0];
    const int d = (int)sd[1];

    // tiled element offset of (node, col0): (nd>>4)*256 + (nd&7)*32 + ((nd>>3)&1)*16
    const int as = ((s >> 4) << 8) + ((s & 7) << 5) + (((s >> 3) & 1) << 4);
    const int ad = ((d >> 4) << 8) + ((d & 7) << 5) + (((d >> 3) & 1) << 4) + 4;

    const v4f ps = *(const v4f*)(p + as);        // h[src] @ W_top (cols 0..3)
    const v4f pd = *(const v4f*)(p + ad);        // h[dst] @ W_bot (cols 4..7)

    v4f o;
    o[0] = ps[0] + pd[0] + b[0];
    o[1] = ps[1] + pd[1] + b[1];
    o[2] = ps[2] + pd[2] + b[2];
    o[3] = ps[3] + pd[3] + b[3];
    __builtin_nontemporal_store(o, (v4f*)(out + (size_t)e * 4));
}

// ---------------------------------------------------------------------------
extern "C" void kernel_launch(void* const* d_in, const int* in_sizes, int n_in,
                              void* d_out, int out_size, void* d_ws, size_t ws_size,
                              hipStream_t stream)
{
    const float*     x   = (const float*)d_in[0];       // [N, 128] f32
    const long long* idx = (const long long*)d_in[1];   // [E, 2]   i64
    const float*     W   = (const float*)d_in[2];       // [256, 4] f32
    const float*     b   = (const float*)d_in[3];       // [4]      f32

    const int nNodes = in_sizes[0] / DIM;
    const int nEdges = in_sizes[1] / 2;

    float* p = (float*)d_ws;                            // nTiles*256 floats (6.4 MB)

    const int nTiles  = (nNodes + 15) / 16;
    const int blocks1 = (nTiles + 7) / 8;               // 8 waves (tiles) per block
    node_logmap_gemm<<<blocks1, 256, 0, stream>>>(x, W, p, nNodes, nTiles);

    const int blocks2 = (nEdges + 255) / 256;
    edge_decode<<<blocks2, 256, 0, stream>>>(p, idx, b, (float*)d_out, nEdges);
}
